// AttentionOCR_5746666242217
// MI455X (gfx1250) — matrix operand order repaired and checked by reading, not compiled, not run on hardware
//
#include <hip/hip_runtime.h>

// ---------------------------------------------------------------------------
// AttentionOCR self-attention, fused flash-attention style, CDNA5 (gfx1250).
// B=4, C=256, N=4096, CQK=32.  bf16 WMMA (V_WMMA_F32_16X16X32_BF16), f32 acc.
// WMMA operands fetched with per-lane B128 loads matching the register layout.
// V blocks streamed into double-buffered LDS via the Tensor Data Mover.
// Single-wave workgroups: wave-synchronous LDS (no barriers needed).
// ---------------------------------------------------------------------------

#define BATCH   4
#define CDIM    256
#define CQK_D   32
#define NPOS    4096

typedef __attribute__((ext_vector_type(16))) __bf16        v16bf;
typedef __attribute__((ext_vector_type(8)))  float         v8f;
typedef __attribute__((ext_vector_type(4)))  unsigned int  u32x4;
typedef __attribute__((ext_vector_type(8)))  int           i32x8;
typedef __attribute__((ext_vector_type(4)))  int           i32x4;

union BF16Vec {
    unsigned int u[8];
    u32x4        q[2];
    v16bf        v;
};

// ---- f32 -> bf16 (packed) --------------------------------------------------
#if __has_builtin(__builtin_amdgcn_cvt_pk_bf16_f32)
#warning "CDNA5 probe: using __builtin_amdgcn_cvt_pk_bf16_f32"
__device__ __forceinline__ unsigned int f2bf2(float lo, float hi) {
    auto r = __builtin_amdgcn_cvt_pk_bf16_f32(lo, hi);
    return __builtin_bit_cast(unsigned int, r);
}
#else
#warning "CDNA5 probe: cvt_pk_bf16 builtin absent; using inline-asm v_cvt_pk_bf16_f32"
__device__ __forceinline__ unsigned int f2bf2(float lo, float hi) {
    unsigned int r;
    asm("v_cvt_pk_bf16_f32 %0, %1, %2" : "=v"(r) : "v"(lo), "v"(hi));
    return r;
}
#endif

__device__ __forceinline__ unsigned short f2bf(float f) {
    return (unsigned short)f2bf2(f, f);
}

// ---- Tensor Data Mover: async strided 2D tile load Global -> LDS -----------
#if __has_builtin(__builtin_amdgcn_tensor_load_to_lds) && \
    __has_builtin(__builtin_amdgcn_s_wait_tensorcnt)
#define HAVE_TDM 1
#warning "CDNA5 probe: TDM path ACTIVE (tensor_load_to_lds + s_wait_tensorcnt)"
#else
#warning "CDNA5 probe: TDM builtins ABSENT; using vector-copy fallback"
#endif

#ifdef HAVE_TDM
// Load a [256 rows x 32 elem] bf16 tile, row stride 4096 elem, to LDS
// (packed rows of 64B -> LDS layout [ch][key]).
__device__ __forceinline__ void tdm_load_v(const unsigned short* gsrc,
                                           unsigned int lds_byte_off) {
    unsigned long long ga = (unsigned long long)(size_t)gsrc;
    u32x4 g0;
    g0[0] = 1u;                                    // count=1, user descriptor
    g0[1] = lds_byte_off;                          // lds_addr (bytes)
    g0[2] = (unsigned int)(ga & 0xFFFFFFFFu);      // global_addr[31:0]
    g0[3] = (unsigned int)((ga >> 32) & 0x01FFFFFFu) | (2u << 30); // type=2
    i32x8 g1;
    g1[0] = 0x00010000;          // data_size=1 (2B), no multicast/pad/iterate
    g1[1] = (int)(4096u << 16);  // tensor_dim0 = 4096 (bits 79:48, low half)
    g1[2] = (int)(256u << 16);   // dim0 hi=0 | tensor_dim1 = 256 (low half)
    g1[3] = (int)(32u << 16);    // dim1 hi=0 | tile_dim0 = 32
    g1[4] = 256;                 // tile_dim1 = 256, tile_dim2 = 0
    g1[5] = 4096;                // tensor_dim0_stride = 4096 (48-bit, lo dword)
    g1[6] = 0;                   // stride hi | tensor_dim1_stride lo
    g1[7] = 0;
    i32x4 z  = (i32x4){0, 0, 0, 0};
#if __clang_major__ >= 23
    i32x8 z8 = (i32x8){0, 0, 0, 0, 0, 0, 0, 0};
    __builtin_amdgcn_tensor_load_to_lds(g0, g1, z, z, z8, 0);
#else
    __builtin_amdgcn_tensor_load_to_lds(g0, g1, z, z, 0);
#endif
}
#endif

// Fallback staging: each lane copies 8 channel rows of 64B.
__device__ __forceinline__ void stage_v_copy(const unsigned short* gsrc,
                                             unsigned short* dst, int lane) {
#pragma unroll
    for (int rr = 0; rr < 8; ++rr) {
        int ch = lane * 8 + rr;
        const u32x4* s = (const u32x4*)(gsrc + (size_t)ch * NPOS);
        u32x4*       d = (u32x4*)(dst + ch * 32);
        d[0] = s[0]; d[1] = s[1]; d[2] = s[2]; d[3] = s[3];
    }
}

// ---------------------------------------------------------------------------
// Kernel 1: QKV projections (1x1 convs) -> bf16 Q[N,32], K[N,32], V[C,N]
// One wave per 16x16 output tile.  20 col-tiles: 2 Q, 2 K, 16 V.
// ---------------------------------------------------------------------------
__global__ __launch_bounds__(32)
void qkv_proj_kernel(const float* __restrict__ x,
                     const float* __restrict__ w1, const float* __restrict__ b1,
                     const float* __restrict__ w2, const float* __restrict__ b2,
                     const float* __restrict__ w3, const float* __restrict__ b3,
                     unsigned short* __restrict__ Qst,
                     unsigned short* __restrict__ Kst,
                     unsigned short* __restrict__ Vst) {
    __shared__ unsigned short ldsX[16 * 32];   // x^T tile, [pos][ch], bf16

    const int bid     = blockIdx.x;
    const int t       = bid % 20;
    const int rowTile = (bid / 20) % (NPOS / 16);
    const int b       = bid / (20 * (NPOS / 16));
    const int n0      = rowTile * 16;

    const float* w; const float* bias; int o0;
    bool isV = false; unsigned short* outp;
    if (t < 2)       { w = w1; bias = b1; outp = Qst; o0 = t * 16;       }
    else if (t < 4)  { w = w2; bias = b2; outp = Kst; o0 = (t - 2) * 16; }
    else             { w = w3; bias = b3; outp = Vst; o0 = (t - 4) * 16; isV = true; }

    const int lane = threadIdx.x;
    const int l16  = lane & 15;
    const int hi   = lane >> 4;

    v8f acc = {};

    for (int c0 = 0; c0 < CDIM; c0 += 32) {
        // ---- stage A = x^T[16 pos][32 ch] into LDS as bf16 (coalesced) ----
        const float* xrow = x + ((size_t)(b * CDIM + c0 + lane)) * NPOS + n0;
        const float4* xrow4 = (const float4*)xrow;
#pragma unroll
        for (int p4 = 0; p4 < 4; ++p4) {
            float4 xv = xrow4[p4];
            ldsX[(p4 * 4 + 0) * 32 + lane] = f2bf(xv.x);
            ldsX[(p4 * 4 + 1) * 32 + lane] = f2bf(xv.y);
            ldsX[(p4 * 4 + 2) * 32 + lane] = f2bf(xv.z);
            ldsX[(p4 * 4 + 3) * 32 + lane] = f2bf(xv.w);
        }
        // single wave32 workgroup: DS ops are in-order, no barrier needed

        // ---- A operand: row m = l16; K-halves contiguous -> 2x ds_load_b128
        BF16Vec A;
        const unsigned short* xa = &ldsX[l16 * 32 + 16 * hi];
        A.q[0] = *(const u32x4*)xa;          // K = 16hi .. 16hi+7
        A.q[1] = *(const u32x4*)(xa + 8);    // K = 16hi+8 .. 16hi+15

        // ---- B operand: col n = l16; K = 16hi..16hi+15 contiguous in w ----
        BF16Vec Bv;
        const float4* wr =
            (const float4*)(w + (size_t)(o0 + l16) * CDIM + c0 + 16 * hi);
        float4 f0 = wr[0], f1 = wr[1], f2c = wr[2], f3 = wr[3];
        Bv.u[0] = f2bf2(f0.x, f0.y);  Bv.u[1] = f2bf2(f0.z, f0.w);
        Bv.u[2] = f2bf2(f1.x, f1.y);  Bv.u[3] = f2bf2(f1.z, f1.w);
        Bv.u[4] = f2bf2(f2c.x, f2c.y); Bv.u[5] = f2bf2(f2c.z, f2c.w);
        Bv.u[6] = f2bf2(f3.x, f3.y);  Bv.u[7] = f2bf2(f3.z, f3.w);

        acc = __builtin_amdgcn_wmma_f32_16x16x32_bf16(
                  false, A.v, false, Bv.v, (short)0, acc, false, false);
    }

    // D[m=pos][n=out] ; lane col n = l16, rows m = r + 8*hi.  Add bias, store.
    const float bb = bias[o0 + l16];
    if (isV) {
        // V is channel-major [B][C][N]: lane's 8 positions are contiguous.
        unsigned int* dst = (unsigned int*)
            (Vst + ((size_t)(b * CDIM + o0 + l16)) * NPOS + n0 + 8 * hi);
#pragma unroll
        for (int rr = 0; rr < 4; ++rr)
            dst[rr] = f2bf2(acc[2 * rr] + bb, acc[2 * rr + 1] + bb);
    } else {
#pragma unroll
        for (int r = 0; r < 8; ++r) {
            int m = r + 8 * hi;
            outp[((size_t)(b * NPOS) + n0 + m) * CQK_D + o0 + l16] =
                f2bf(acc[r] + bb);
        }
    }
}

// ---------------------------------------------------------------------------
// Kernel 2: flash attention + gamma*out + x.  One wave per 16-query tile.
//   S^T = K_blk(16x32ch) @ Q^T(32ch x 16q)      -> softmax axis in VGPR rows
//   O^T = V^T(16ch x 32key) @ P^T(32key x 16q)  -> 16 accumulator tiles
// V blocks double-buffered in LDS via TDM (tensor_load_to_lds + TENSORcnt).
// ---------------------------------------------------------------------------
__global__ __launch_bounds__(32)
void flash_attn_kernel(const unsigned short* __restrict__ Qst,
                       const unsigned short* __restrict__ Kst,
                       const unsigned short* __restrict__ Vst,
                       const float* __restrict__ x,
                       const float* __restrict__ gamma_p,
                       float* __restrict__ out) {
    __shared__ unsigned short ldsV[2][CDIM * 32];  // [buf][ch][key] bf16, 2x16KB
    __shared__ unsigned short ldsP[16 * 32];       // [query][key]   bf16, 1KB

    const int bid     = blockIdx.x;
    const int rowTile = bid % (NPOS / 16);
    const int b       = bid / (NPOS / 16);
    const int n0      = rowTile * 16;

    const int lane = threadIdx.x;
    const int l16  = lane & 15;
    const int hi   = lane >> 4;

    const unsigned short* vbase = Vst + (size_t)(b * CDIM) * NPOS;  // [ch][key]

    // Persistent B operand: Q^T (32 ch x 16 queries); K contiguous per half.
    BF16Vec Qb;
    {
        const unsigned short* qrow =
            Qst + ((size_t)(b * NPOS) + n0 + l16) * CQK_D + 16 * hi;
        Qb.q[0] = *(const u32x4*)qrow;
        Qb.q[1] = *(const u32x4*)(qrow + 8);
    }

    v8f O[16];
#pragma unroll
    for (int c = 0; c < 16; ++c) O[c] = (v8f){};

    float m_run = -1e30f;
    float l_run = 0.0f;

    // ---- prologue: start streaming V block 0 into buffer 0 ----------------
#ifdef HAVE_TDM
    tdm_load_v(vbase, (unsigned int)(size_t)&ldsV[0][0]);
#else
    stage_v_copy(vbase, &ldsV[0][0], lane);
#endif

    for (int j0 = 0; j0 < NPOS; j0 += 32) {
        const int cur = (j0 >> 5) & 1;

        // ---- kick off async TDM for the NEXT V block into the other buffer
        if (j0 + 32 < NPOS) {
#ifdef HAVE_TDM
            tdm_load_v(vbase + (j0 + 32),
                       (unsigned int)(size_t)&ldsV[cur ^ 1][0]);
#else
            stage_v_copy(vbase + (j0 + 32), &ldsV[cur ^ 1][0], lane);
#endif
            // prefetch next K block while we are at it
            __builtin_prefetch(Kst + ((size_t)(b * NPOS) + j0 + 32 + lane) * CQK_D, 0, 3);
        }

        // ---- S^T tiles: A = K rows (key-major), B = Q^T -------------------
        float s[2][8];
#pragma unroll
        for (int t = 0; t < 2; ++t) {
            BF16Vec Ka;
            const unsigned short* krow =
                Kst + ((size_t)(b * NPOS) + j0 + 16 * t + l16) * CQK_D;
            Ka.q[0] = *(const u32x4*)(krow + 16 * hi);
            Ka.q[1] = *(const u32x4*)(krow + 16 * hi + 8);
            v8f sv = {};
            sv = __builtin_amdgcn_wmma_f32_16x16x32_bf16(
                     false, Ka.v, false, Qb.v, (short)0, sv, false, false);
#pragma unroll
            for (int r = 0; r < 8; ++r) s[t][r] = sv[r];
        }

        // ---- online softmax over keys (rows) for this 32-key block --------
        float mx = -1e30f;
#pragma unroll
        for (int t = 0; t < 2; ++t)
#pragma unroll
            for (int r = 0; r < 8; ++r) mx = fmaxf(mx, s[t][r]);
        mx = fmaxf(mx, __shfl_xor(mx, 16));
        const float m_new = fmaxf(m_run, mx);

        float sum = 0.0f;
#pragma unroll
        for (int t = 0; t < 2; ++t)
#pragma unroll
            for (int r = 0; r < 8; ++r) {
                float p = __expf(s[t][r] - m_new);
                s[t][r] = p;
                sum += p;
            }
        sum += __shfl_xor(sum, 16);

        const float alpha = __expf(m_run - m_new);
        l_run = l_run * alpha + sum;
        m_run = m_new;
#pragma unroll
        for (int c = 0; c < 16; ++c) O[c] = O[c] * alpha;

        // ---- write P^T to LDS [query][key]: 8 consecutive keys per lane ---
#pragma unroll
        for (int t = 0; t < 2; ++t) {
            u32x4 pv;
            pv[0] = f2bf2(s[t][0], s[t][1]);
            pv[1] = f2bf2(s[t][2], s[t][3]);
            pv[2] = f2bf2(s[t][4], s[t][5]);
            pv[3] = f2bf2(s[t][6], s[t][7]);
            *(u32x4*)&ldsP[l16 * 32 + 16 * t + 8 * hi] = pv;
        }

        // ---- wait for the CURRENT V buffer (allow next to stay in flight) -
#ifdef HAVE_TDM
        if (j0 + 32 < NPOS) __builtin_amdgcn_s_wait_tensorcnt(1);
        else                __builtin_amdgcn_s_wait_tensorcnt(0);
#endif

        // ---- B operand P^T: K = 16hi..16hi+15 contiguous ------------------
        BF16Vec Pb;
        const unsigned short* pr = &ldsP[l16 * 32 + 16 * hi];
        Pb.q[0] = *(const u32x4*)pr;
        Pb.q[1] = *(const u32x4*)(pr + 8);

        // ---- O^T += V^T @ P^T : 16 channel tiles --------------------------
        const unsigned short* vb = &ldsV[cur][0];
#pragma unroll
        for (int c = 0; c < 16; ++c) {
            BF16Vec Va;
            const unsigned short* vrow = &vb[(c * 16 + l16) * 32];
            Va.q[0] = *(const u32x4*)(vrow + 16 * hi);
            Va.q[1] = *(const u32x4*)(vrow + 16 * hi + 8);
            O[c] = __builtin_amdgcn_wmma_f32_16x16x32_bf16(
                       false, Va.v, false, Pb.v, (short)0, O[c], false, false);
        }
    }

    // ---- epilogue: out = gamma * (O / l) + x ------------------------------
    const float inv_l = 1.0f / l_run;
    const float g     = gamma_p[0];
#pragma unroll
    for (int c = 0; c < 16; ++c)
#pragma unroll
        for (int r = 0; r < 8; ++r) {
            int ch = c * 16 + r + 8 * hi;                 // D rows = channels
            size_t idx = ((size_t)(b * CDIM + ch)) * NPOS + n0 + l16;
            out[idx] = g * (O[c][r] * inv_l) + x[idx];
        }
}

// ---------------------------------------------------------------------------
extern "C" void kernel_launch(void* const* d_in, const int* in_sizes, int n_in,
                              void* d_out, int out_size, void* d_ws, size_t ws_size,
                              hipStream_t stream) {
    const float* x     = (const float*)d_in[0];
    const float* w1    = (const float*)d_in[1];
    const float* b1    = (const float*)d_in[2];
    const float* w2    = (const float*)d_in[3];
    const float* b2    = (const float*)d_in[4];
    const float* w3    = (const float*)d_in[5];
    const float* b3    = (const float*)d_in[6];
    const float* gamma = (const float*)d_in[7];
    float* out = (float*)d_out;

    unsigned short* Qst = (unsigned short*)d_ws;                 // [B,N,32]  bf16
    unsigned short* Kst = Qst + (size_t)BATCH * NPOS * CQK_D;    // [B,N,32]  bf16
    unsigned short* Vst = Kst + (size_t)BATCH * NPOS * CQK_D;    // [B,C,N]   bf16

    // 20 column tiles (2 Q, 2 K, 16 V) x 256 row tiles x 4 batches
    dim3 grid1(BATCH * (NPOS / 16) * 20);
    qkv_proj_kernel<<<grid1, 32, 0, stream>>>(x, w1, b1, w2, b2, w3, b3,
                                              Qst, Kst, Vst);

    dim3 grid2(BATCH * (NPOS / 16));
    flash_attn_kernel<<<grid2, 32, 0, stream>>>(Qst, Kst, Vst, x, gamma, out);
}
